// TransformerAttention_69389491634301
// MI455X (gfx1250) — matrix operand order repaired and checked by reading, not compiled, not run on hardware
//
#include <hip/hip_runtime.h>

typedef _Float16 half_t;
typedef half_t v16h __attribute__((ext_vector_type(16)));
typedef half_t v8h  __attribute__((ext_vector_type(8)));
typedef float  v8f  __attribute__((ext_vector_type(8)));

#define DMODEL 1024
#define NHEAD  16
#define HDIM   64
#define SEQ    2048
#define BATCH  4
#define BS     (BATCH * SEQ)   // 8192 rows

static __device__ inline v16h make_v16(v8h lo, v8h hi) {
    v16h r;
#pragma unroll
    for (int i = 0; i < 8; ++i) { r[i] = lo[i]; r[i + 8] = hi[i]; }
    return r;
}

// Flat (generic) pointers to LDS carry the wave-relative LDS byte offset in
// the low 32 bits (aperture mapping), which is exactly the VDST operand of
// the async-to-LDS instructions.
static __device__ inline unsigned lds_off_of(const void* p) {
    return (unsigned)(unsigned long long)p;
}

// CDNA5 async copy: 16B global -> LDS, tracked by ASYNCcnt (no VGPR data).
static __device__ inline void async_copy16(unsigned lds_off, const void* gptr) {
    asm volatile("global_load_async_to_lds_b128 %0, %1, off"
                 :: "v"(lds_off), "v"((unsigned long long)(size_t)gptr)
                 : "memory");
}
static __device__ inline void wait_async0() {
    asm volatile("s_wait_asynccnt 0" ::: "memory");
}

// ---------------------------------------------------------------------------
// Stage 0: f32 -> f16 conversion (x and weight matrices)
// ---------------------------------------------------------------------------
__global__ void f32_to_f16_kernel(const float* __restrict__ in,
                                  half_t* __restrict__ out, int n) {
    int i = blockIdx.x * blockDim.x + threadIdx.x;
    if (i < n) out[i] = (half_t)in[i];
}

// ---------------------------------------------------------------------------
// Stage 1: fused QKV projection.  y = x @ W^T + b
// Workgroup: 256 thr = 8 waves, tile 256(M) x 64(N); each wave 32x64
// (2 A-frags x 4 B-frags = 8 WMMAs / k-step).  B tile double-buffered in LDS
// via GLOBAL_LOAD_ASYNC_TO_LDS_B128 (ASYNCcnt), prefetching k+32 while
// computing k.
// z=0 -> Q [B,H,S,HD], z=1 -> K [B,H,S,HD], z=2 -> V^T [B,H,HD,S]
// ---------------------------------------------------------------------------
__global__ __launch_bounds__(256) void gemm_qkv_kernel(
    const half_t* __restrict__ xh,
    const half_t* __restrict__ Wq, const half_t* __restrict__ Wk,
    const half_t* __restrict__ Wv,
    const float* __restrict__ bq, const float* __restrict__ bk,
    const float* __restrict__ bv,
    half_t* __restrict__ Qh, half_t* __restrict__ Kh, half_t* __restrict__ Vt)
{
    const int z    = blockIdx.z;
    const half_t* W    = (z == 0) ? Wq : (z == 1) ? Wk : Wv;
    const float*  bias = (z == 0) ? bq : (z == 1) ? bk : bv;

    const int m0   = blockIdx.x * 256;
    const int n0   = blockIdx.y * 64;
    const int tid  = threadIdx.x;
    const int wave = tid >> 5;
    const int lane = tid & 31;
    const int l15  = lane & 15;
    const int hi   = lane >> 4;

    __shared__ half_t Bs[2][64][40];      // double-buffered 64N x 32K tile

    v8f acc[2][4] = {};

    const int     rbase = m0 + wave * 32;
    const half_t* Arow0 = xh + (size_t)(rbase + l15) * DMODEL;
    const half_t* Arow1 = Arow0 + (size_t)16 * DMODEL;
    const int     aoff  = hi * 8;

    // per-thread staging slot: row sr, 8-half chunk at scol
    const int sr = tid >> 2, scol = (tid & 3) * 8;
    const half_t* Wrow = W + (size_t)(n0 + sr) * DMODEL + scol;

    // prologue: async-stage k=0 tile into buffer 0
    async_copy16(lds_off_of(&Bs[0][sr][scol]), Wrow);
    wait_async0();
    __syncthreads();

    int buf = 0;
    for (int k = 0; k < DMODEL; k += 32) {
        // prefetch next B tile into the other buffer (overlaps with WMMAs)
        if (k + 32 < DMODEL)
            async_copy16(lds_off_of(&Bs[buf ^ 1][sr][scol]), Wrow + k + 32);

        v16h a0 = make_v16(*reinterpret_cast<const v8h*>(Arow0 + k + aoff),
                           *reinterpret_cast<const v8h*>(Arow0 + k + 16 + aoff));
        v16h a1 = make_v16(*reinterpret_cast<const v8h*>(Arow1 + k + aoff),
                           *reinterpret_cast<const v8h*>(Arow1 + k + 16 + aoff));
#pragma unroll
        for (int f = 0; f < 4; ++f) {
            int bn   = f * 16 + l15;
            int koff = hi * 8;
            v16h b = make_v16(*reinterpret_cast<const v8h*>(&Bs[buf][bn][koff]),
                              *reinterpret_cast<const v8h*>(&Bs[buf][bn][koff + 16]));
            acc[0][f] = __builtin_amdgcn_wmma_f32_16x16x32_f16(
                false, a0, false, b, (short)0, acc[0][f], false, false);
            acc[1][f] = __builtin_amdgcn_wmma_f32_16x16x32_f16(
                false, a1, false, b, (short)0, acc[1][f], false, false);
        }
        wait_async0();
        __syncthreads();
        buf ^= 1;
    }

    // epilogue: bias + scatter into head layout
#pragma unroll
    for (int ph = 0; ph < 2; ++ph)
#pragma unroll
        for (int f = 0; f < 4; ++f) {
            int   n  = n0 + f * 16 + l15;
            float bn = bias[n];
            int   h  = n >> 6, d = n & 63;
#pragma unroll
            for (int r = 0; r < 8; ++r) {
                int   m = rbase + ph * 16 + r + hi * 8;
                int   b = m >> 11, s = m & 2047;
                float v = acc[ph][f][r] + bn;
                if (z == 2) {
                    Vt[((size_t)((b * NHEAD + h) * HDIM + d)) * SEQ + s] = (half_t)v;
                } else {
                    half_t* dst = (z == 0) ? Qh : Kh;
                    dst[((size_t)((b * NHEAD + h) * SEQ + s)) * HDIM + d] = (half_t)v;
                }
            }
        }
}

// ---------------------------------------------------------------------------
// Stage 2: flash attention per (b,h).  grid = (S/64, B*H), 128 thr = 4 waves.
// K/V tiles double-buffered in LDS via async-to-LDS; next key-tile prefetch
// is issued before the current tile's WMMAs/softmax.
// ---------------------------------------------------------------------------
__global__ __launch_bounds__(128) void attn_kernel(
    const half_t* __restrict__ Qh, const half_t* __restrict__ Kh,
    const half_t* __restrict__ Vt, half_t* __restrict__ Ctx)
{
    const int qt   = blockIdx.x;          // 0..31
    const int bh   = blockIdx.y;          // 0..63 (= b*16 + h)
    const int tid  = threadIdx.x;
    const int wave = tid >> 5;
    const int lane = tid & 31;
    const int l15  = lane & 15;
    const int hi   = lane >> 4;

    __shared__ half_t Ks[2][64][72];      // keys x d (+pad), double-buffered
    __shared__ half_t Vs[2][64][72];      // d x keys (+pad), double-buffered
    __shared__ half_t Ps[4][16][72];      // per-wave P tile 16 x 64 (+pad)

    const half_t* Qbase = Qh + (size_t)bh * SEQ * HDIM;
    const half_t* Kbase = Kh + (size_t)bh * SEQ * HDIM;
    const half_t* Vbase = Vt + (size_t)bh * HDIM * SEQ;

    // staging slots: 4 chunks per thread per tile
    int srow[4], scol4[4];
#pragma unroll
    for (int i = 0; i < 4; ++i) {
        int c = tid + i * 128;            // 0..511
        srow[i]  = c >> 3;
        scol4[i] = (c & 7) * 8;
    }

    // preload this wave's Q fragments (16 rows x 64 d -> two K=32 steps)
    const int qrow = qt * 64 + wave * 16 + l15;
    v16h qfrag[2];
#pragma unroll
    for (int ks = 0; ks < 2; ++ks) {
        const half_t* qp = Qbase + (size_t)qrow * HDIM + ks * 32;
        qfrag[ks] = make_v16(*reinterpret_cast<const v8h*>(qp + hi * 8),
                             *reinterpret_cast<const v8h*>(qp + 16 + hi * 8));
    }

    float mstate[8], lstate[8];
#pragma unroll
    for (int r = 0; r < 8; ++r) { mstate[r] = -3.0e38f; lstate[r] = 0.0f; }
    v8f ctxacc[4] = {};

    // prologue: async-stage tile kt=0 into buffer 0
#pragma unroll
    for (int i = 0; i < 4; ++i) {
        async_copy16(lds_off_of(&Ks[0][srow[i]][scol4[i]]),
                     &Kbase[(size_t)srow[i] * HDIM + scol4[i]]);
        async_copy16(lds_off_of(&Vs[0][srow[i]][scol4[i]]),
                     &Vbase[(size_t)srow[i] * SEQ + scol4[i]]);
    }
    wait_async0();
    __syncthreads();

    int buf = 0;
    for (int kt = 0; kt < SEQ / 64; ++kt) {
        // prefetch next key tile into the other buffer
        if (kt + 1 < SEQ / 64) {
            const int s1 = (kt + 1) * 64;
#pragma unroll
            for (int i = 0; i < 4; ++i) {
                async_copy16(lds_off_of(&Ks[buf ^ 1][srow[i]][scol4[i]]),
                             &Kbase[(size_t)(s1 + srow[i]) * HDIM + scol4[i]]);
                async_copy16(lds_off_of(&Vs[buf ^ 1][srow[i]][scol4[i]]),
                             &Vbase[(size_t)srow[i] * SEQ + s1 + scol4[i]]);
            }
        }

        // scores: S = Q K^T  (M=16 q-rows, N=64 keys, K=64 d)
        v8f sc[4] = {};
#pragma unroll
        for (int f = 0; f < 4; ++f) {
#pragma unroll
            for (int ks = 0; ks < 2; ++ks) {
                int brow = f * 16 + l15;
                int koff = ks * 32 + hi * 8;
                v16h b = make_v16(
                    *reinterpret_cast<const v8h*>(&Ks[buf][brow][koff]),
                    *reinterpret_cast<const v8h*>(&Ks[buf][brow][koff + 16]));
                sc[f] = __builtin_amdgcn_wmma_f32_16x16x32_f16(
                    false, qfrag[ks], false, b, (short)0, sc[f], false, false);
            }
        }

        // online softmax (rows live in VGPR r at lanes of one 16-lane half)
        float alpha[8];
#pragma unroll
        for (int r = 0; r < 8; ++r) {
            float mx = -3.0e38f;
#pragma unroll
            for (int f = 0; f < 4; ++f) {
                sc[f][r] *= 0.125f;                 // 1/sqrt(64)
                mx = fmaxf(mx, sc[f][r]);
            }
#pragma unroll
            for (int off = 1; off < 16; off <<= 1)
                mx = fmaxf(mx, __shfl_xor(mx, off, 32));
            float mnew = fmaxf(mstate[r], mx);
            alpha[r]   = __expf(mstate[r] - mnew);
            mstate[r]  = mnew;

            float s = 0.0f;
#pragma unroll
            for (int f = 0; f < 4; ++f) {
                float p = __expf(sc[f][r] - mnew);
                sc[f][r] = p;
                s += p;
            }
#pragma unroll
            for (int off = 1; off < 16; off <<= 1)
                s += __shfl_xor(s, off, 32);
            lstate[r] = lstate[r] * alpha[r] + s;
        }
#pragma unroll
        for (int f = 0; f < 4; ++f)
#pragma unroll
            for (int r = 0; r < 8; ++r)
                ctxacc[f][r] *= alpha[r];

        // reshape P: accumulator layout -> A layout via LDS
#pragma unroll
        for (int f = 0; f < 4; ++f)
#pragma unroll
            for (int r = 0; r < 8; ++r)
                Ps[wave][r + hi * 8][f * 16 + l15] = (half_t)sc[f][r];
        __syncthreads();

        // ctx += P V   (M=16 q-rows, N=64 d, K=64 keys)
#pragma unroll
        for (int ks = 0; ks < 2; ++ks) {
            int kk0 = ks * 32;
            v16h pa = make_v16(
                *reinterpret_cast<const v8h*>(&Ps[wave][l15][kk0 + hi * 8]),
                *reinterpret_cast<const v8h*>(&Ps[wave][l15][kk0 + 16 + hi * 8]));
#pragma unroll
            for (int f = 0; f < 4; ++f) {
                int vrow = f * 16 + l15;
                int koff = kk0 + hi * 8;
                v16h vb = make_v16(
                    *reinterpret_cast<const v8h*>(&Vs[buf][vrow][koff]),
                    *reinterpret_cast<const v8h*>(&Vs[buf][vrow][koff + 16]));
                ctxacc[f] = __builtin_amdgcn_wmma_f32_16x16x32_f16(
                    false, pa, false, vb, (short)0, ctxacc[f], false, false);
            }
        }
        wait_async0();
        __syncthreads();
        buf ^= 1;
    }

    // finalize: ctx /= l, store to [BS, D] with col = h*64 + d
    const int b = bh >> 4, h = bh & 15;
#pragma unroll
    for (int f = 0; f < 4; ++f)
#pragma unroll
        for (int r = 0; r < 8; ++r) {
            float v  = ctxacc[f][r] / lstate[r];
            int   m  = b * SEQ + qt * 64 + wave * 16 + r + hi * 8;
            int   cl = h * HDIM + f * 16 + l15;
            Ctx[(size_t)m * DMODEL + cl] = (half_t)v;
        }
}

// ---------------------------------------------------------------------------
// Stage 3: output projection.  out = Ctx @ Wo^T + bo  (f32 output)
// Same tiling as stage 1 (32x64 per wave, async double-buffered B tile).
// ---------------------------------------------------------------------------
__global__ __launch_bounds__(256) void gemm_out_kernel(
    const half_t* __restrict__ Ctx, const half_t* __restrict__ Wo,
    const float* __restrict__ bo, float* __restrict__ out)
{
    const int m0   = blockIdx.x * 256;
    const int n0   = blockIdx.y * 64;
    const int tid  = threadIdx.x;
    const int wave = tid >> 5;
    const int lane = tid & 31;
    const int l15  = lane & 15;
    const int hi   = lane >> 4;

    __shared__ half_t Bs[2][64][40];

    v8f acc[2][4] = {};
    const int     rbase = m0 + wave * 32;
    const half_t* Arow0 = Ctx + (size_t)(rbase + l15) * DMODEL;
    const half_t* Arow1 = Arow0 + (size_t)16 * DMODEL;
    const int     aoff  = hi * 8;

    const int sr = tid >> 2, scol = (tid & 3) * 8;
    const half_t* Wrow = Wo + (size_t)(n0 + sr) * DMODEL + scol;

    async_copy16(lds_off_of(&Bs[0][sr][scol]), Wrow);
    wait_async0();
    __syncthreads();

    int buf = 0;
    for (int k = 0; k < DMODEL; k += 32) {
        if (k + 32 < DMODEL)
            async_copy16(lds_off_of(&Bs[buf ^ 1][sr][scol]), Wrow + k + 32);

        v16h a0 = make_v16(*reinterpret_cast<const v8h*>(Arow0 + k + aoff),
                           *reinterpret_cast<const v8h*>(Arow0 + k + 16 + aoff));
        v16h a1 = make_v16(*reinterpret_cast<const v8h*>(Arow1 + k + aoff),
                           *reinterpret_cast<const v8h*>(Arow1 + k + 16 + aoff));
#pragma unroll
        for (int f = 0; f < 4; ++f) {
            int koff = hi * 8;
            v16h b = make_v16(
                *reinterpret_cast<const v8h*>(&Bs[buf][f * 16 + l15][koff]),
                *reinterpret_cast<const v8h*>(&Bs[buf][f * 16 + l15][koff + 16]));
            acc[0][f] = __builtin_amdgcn_wmma_f32_16x16x32_f16(
                false, a0, false, b, (short)0, acc[0][f], false, false);
            acc[1][f] = __builtin_amdgcn_wmma_f32_16x16x32_f16(
                false, a1, false, b, (short)0, acc[1][f], false, false);
        }
        wait_async0();
        __syncthreads();
        buf ^= 1;
    }

#pragma unroll
    for (int ph = 0; ph < 2; ++ph)
#pragma unroll
        for (int f = 0; f < 4; ++f) {
            int   n  = n0 + f * 16 + l15;
            float bn = bo[n];
#pragma unroll
            for (int r = 0; r < 8; ++r) {
                int m = rbase + ph * 16 + r + hi * 8;
                out[(size_t)m * DMODEL + n] = acc[ph][f][r] + bn;
            }
        }
}

// ---------------------------------------------------------------------------
extern "C" void kernel_launch(void* const* d_in, const int* in_sizes, int n_in,
                              void* d_out, int out_size, void* d_ws, size_t ws_size,
                              hipStream_t stream) {
    (void)in_sizes; (void)n_in; (void)out_size; (void)ws_size;

    const float* x  = (const float*)d_in[0];
    const float* Wq = (const float*)d_in[1];
    const float* bq = (const float*)d_in[2];
    const float* Wk = (const float*)d_in[3];
    const float* bk = (const float*)d_in[4];
    const float* Wv = (const float*)d_in[5];
    const float* bv = (const float*)d_in[6];
    const float* Wo = (const float*)d_in[7];
    const float* bo = (const float*)d_in[8];
    float* out = (float*)d_out;

    const size_t XH = (size_t)BS * DMODEL;       // 8,388,608
    const size_t WH = (size_t)DMODEL * DMODEL;   // 1,048,576

    half_t* p   = (half_t*)d_ws;
    half_t* xh  = p;
    half_t* wqh = xh  + XH;
    half_t* wkh = wqh + WH;
    half_t* wvh = wkh + WH;
    half_t* woh = wvh + WH;
    half_t* qh  = woh + WH;
    half_t* kh  = qh  + XH;
    half_t* vt  = kh  + XH;
    half_t* ctx = vt  + XH;

    f32_to_f16_kernel<<<(int)(XH / 256), 256, 0, stream>>>(x,  xh,  (int)XH);
    f32_to_f16_kernel<<<(int)(WH / 256), 256, 0, stream>>>(Wq, wqh, (int)WH);
    f32_to_f16_kernel<<<(int)(WH / 256), 256, 0, stream>>>(Wk, wkh, (int)WH);
    f32_to_f16_kernel<<<(int)(WH / 256), 256, 0, stream>>>(Wv, wvh, (int)WH);
    f32_to_f16_kernel<<<(int)(WH / 256), 256, 0, stream>>>(Wo, woh, (int)WH);

    gemm_qkv_kernel<<<dim3(BS / 256, DMODEL / 64, 3), 256, 0, stream>>>(
        xh, wqh, wkh, wvh, bq, bk, bv, qh, kh, vt);

    attn_kernel<<<dim3(SEQ / 64, BATCH * NHEAD), 128, 0, stream>>>(qh, kh, vt, ctx);

    gemm_out_kernel<<<dim3(BS / 256, DMODEL / 64), 256, 0, stream>>>(ctx, woh, bo, out);
}
